// PartitionAttn2_88192858456496
// MI455X (gfx1250) — compile-verified, matrix operand order checked by reading, not run on hardware
//
#include <hip/hip_runtime.h>
#include <stdint.h>

// ---------------------------------------------------------------------------
// PartitionAttn2 (Swin-style window block) for gfx1250 / MI455X.
// Matmuls: v_wmma_f32_16x16x32_f16. GEMM tiles move via async global->LDS
// (GLOBAL_LOAD_ASYNC_TO_LDS_B128, ASYNCcnt) with double buffering.
// ---------------------------------------------------------------------------

typedef __attribute__((ext_vector_type(16))) _Float16 v16h;
typedef __attribute__((ext_vector_type(8)))  float    v8f;
typedef __attribute__((ext_vector_type(4)))  _Float16 h4;

union V16H { v16h v; unsigned u[8]; _Float16 h[16]; };
union V8F  { v8f  v; float f[8]; };

#define BB   32
#define CC   512
#define HH   64
#define WW   64
#define PP64 64           // tokens per window
#define NH   16
#define HD   32
#define ROWS 131072       // B*G*PP = 32*64*64
#define DFF  2048
#define LDA  40           // LDS tile stride in halves (80B: 16B-aligned rows, bank-clean)

__device__ __forceinline__ v8f wmma16(v16h a, v16h b, v8f c) {
    return __builtin_amdgcn_wmma_f32_16x16x32_f16(false, a, false, b, (short)0, c,
                                                  false, false);
}

// A/B fragment from a row-major [r][k] buffer: r = lane&15, k per ISA 16-bit
// A-matrix 16x32 layout (pairs contiguous -> dword loads). Works for A tiles
// and for B tiles stored transposed as [n][k].
__device__ __forceinline__ v16h load_frag_rowk(const _Float16* base, int ld, int lane) {
    V16H f;
    const _Float16* p = base + (size_t)(lane & 15) * ld;
    const int hi8 = (lane >> 4) << 3;
#pragma unroll
    for (int j2 = 0; j2 < 8; ++j2) {
        int k = ((j2 & 4) << 2) + hi8 + ((j2 & 3) << 1);
        f.u[j2] = *reinterpret_cast<const unsigned*>(p + k);
    }
    return f.v;
}

// B fragment from a [k][n] (row-major over k) buffer: n = lane&15.
__device__ __forceinline__ v16h load_frag_kn(const _Float16* base, int ld, int lane) {
    V16H f;
    const int n = lane & 15;
    const int hi8 = (lane >> 4) << 3;
#pragma unroll
    for (int j2 = 0; j2 < 8; ++j2) {
        int k = ((j2 & 4) << 2) + hi8 + ((j2 & 3) << 1);
        f.h[2 * j2]     = base[(size_t)k * ld + n];
        f.h[2 * j2 + 1] = base[(size_t)(k + 1) * ld + n];
    }
    return f.v;
}

__device__ __forceinline__ float gelu_tanh(float x) {
    float x3 = x * x * x;
    return 0.5f * x * (1.f + tanhf(0.7978845608028654f * (x + 0.044715f * x3)));
}

// Async DMA of 16B/lane from global to LDS (tracked by ASYNCcnt).
// lds_byte_off: LDS byte address (low 32 bits of flat shared pointer).
// gptr: 64-bit global address (GV mode, saddr=off).
__device__ __forceinline__ void async_copy_b128(unsigned lds_byte_off, const void* gptr) {
    unsigned long long ga = (unsigned long long)(uintptr_t)gptr;
    asm volatile("global_load_async_to_lds_b128 %0, %1, off"
                 :: "v"(lds_byte_off), "v"(ga) : "memory");
}
__device__ __forceinline__ void wait_async_le4() {
    asm volatile("s_wait_asynccnt 0x4" ::: "memory");
}
__device__ __forceinline__ void wait_async_0() {
    asm volatile("s_wait_asynccnt 0x0" ::: "memory");
}

// ---------------------------------------------------------------------------
// Prep kernels
// ---------------------------------------------------------------------------

// w: [K][N] f32  ->  wt: [N][K] f16   (coalesced writes along k)
__global__ void transpose_to_f16(const float* __restrict__ w, _Float16* __restrict__ wt,
                                 int K, int N) {
    long long idx = (long long)blockIdx.x * 256 + threadIdx.x;
    if (idx < (long long)K * N) {
        int k = (int)(idx % K);
        int n = (int)(idx / K);
        wt[(size_t)n * K + k] = (_Float16)w[(size_t)k * N + n];
    }
}

// bias_full[h][q][k] = table[rel_idx[q*64+k]][h]
__global__ void build_bias(const float* __restrict__ table, const int* __restrict__ rel_idx,
                           float* __restrict__ bias_full) {
    int idx = blockIdx.x * 256 + threadIdx.x;
    if (idx < NH * PP64 * PP64) {
        int kq = idx & 4095;
        int h  = idx >> 12;
        bias_full[idx] = table[rel_idx[kq] * NH + h];
    }
}

// [B,C,H,W] -> [B*G*PP, C] window partition (coalesced reads)
__global__ void partition_kernel(const float* __restrict__ x, float* __restrict__ xw) {
    size_t idx = (size_t)blockIdx.x * 256 + threadIdx.x;
    if (idx < (size_t)BB * CC * HH * WW) {
        int w  = (int)(idx & 63);
        int h  = (int)((idx >> 6) & 63);
        int c  = (int)((idx >> 12) & 511);
        int b  = (int)(idx >> 21);
        int row = ((b * 8 + (h >> 3)) * 8 + (w >> 3)) * 64 + (h & 7) * 8 + (w & 7);
        xw[(size_t)row * CC + c] = x[idx];
    }
}

// [B*G*PP, C] -> [B,C,H,W] (coalesced writes)
__global__ void departition_kernel(const float* __restrict__ xw, float* __restrict__ out) {
    size_t idx = (size_t)blockIdx.x * 256 + threadIdx.x;
    if (idx < (size_t)BB * CC * HH * WW) {
        int w  = (int)(idx & 63);
        int h  = (int)((idx >> 6) & 63);
        int c  = (int)((idx >> 12) & 511);
        int b  = (int)(idx >> 21);
        int row = ((b * 8 + (h >> 3)) * 8 + (w >> 3)) * 64 + (h & 7) * 8 + (w & 7);
        out[idx] = xw[(size_t)row * CC + c];
    }
}

// ---------------------------------------------------------------------------
// LayerNorm: one block (128 threads) per token row of 512, f32 in -> f16 out
// ---------------------------------------------------------------------------
__global__ __launch_bounds__(128)
void ln_kernel(const float* __restrict__ xw, const float* __restrict__ gamma,
               const float* __restrict__ beta, _Float16* __restrict__ h) {
    const int row = blockIdx.x;
    const int tid = threadIdx.x;
    const float4 xv = reinterpret_cast<const float4*>(xw + (size_t)row * CC)[tid];
    float s  = xv.x + xv.y + xv.z + xv.w;
    float s2 = xv.x * xv.x + xv.y * xv.y + xv.z * xv.z + xv.w * xv.w;
#pragma unroll
    for (int m = 16; m >= 1; m >>= 1) {
        s  += __shfl_xor(s,  m, 32);
        s2 += __shfl_xor(s2, m, 32);
    }
    __shared__ float ps[4], ps2[4];
    if ((tid & 31) == 0) { ps[tid >> 5] = s; ps2[tid >> 5] = s2; }
    __syncthreads();
    s  = ps[0] + ps[1] + ps[2] + ps[3];
    s2 = ps2[0] + ps2[1] + ps2[2] + ps2[3];
    const float mu  = s * (1.f / 512.f);
    const float var = s2 * (1.f / 512.f) - mu * mu;
    const float rs  = rsqrtf(var + 1e-5f);
    const float4 g = reinterpret_cast<const float4*>(gamma)[tid];
    const float4 b = reinterpret_cast<const float4*>(beta)[tid];
    h4 hv;
    hv.x = (_Float16)((xv.x - mu) * rs * g.x + b.x);
    hv.y = (_Float16)((xv.y - mu) * rs * g.y + b.y);
    hv.z = (_Float16)((xv.z - mu) * rs * g.z + b.z);
    hv.w = (_Float16)((xv.w - mu) * rs * g.w + b.w);
    *reinterpret_cast<h4*>(h + (size_t)row * CC + tid * 4) = hv;
}

// ---------------------------------------------------------------------------
// Tiled WMMA GEMM: C[M,N] = A[M,K] * B[K,N] (+bias, epilogue variants)
// A row-major f16, BT is B transposed [N][K] f16.
// Block tile 128x128, 8 waves as 4(M) x 2(N), wave tile 32x64.
// K-tiles staged with double-buffered GLOBAL_LOAD_ASYNC_TO_LDS_B128.
// EPI: 0 = bias -> f16 out; 1 = bias + residual(f32) -> f32 out; 2 = bias+GELU -> f16
// ---------------------------------------------------------------------------
template <int KDIM, int EPI>
__global__ __launch_bounds__(256)
void gemm_kernel(const _Float16* __restrict__ A, const _Float16* __restrict__ BT,
                 const float* __restrict__ bias, const float* __restrict__ resid,
                 void* __restrict__ outp, int N) {
    // smem[buf][mat][128*LDA] halves; LDA=40 -> rows 16B-aligned for b128 DMA.
    __shared__ _Float16 smem[2][2][128 * LDA];
    const int m0   = blockIdx.y * 128;
    const int n0   = blockIdx.x * 128;
    const int tid  = threadIdx.x;
    const int lane = tid & 31;
    const int wave = tid >> 5;
    const int wm   = wave >> 1;   // 0..3
    const int wn   = wave & 1;    // 0..1

    const unsigned smem_base = (unsigned)(uintptr_t)(&smem[0][0][0]);

    v8f acc[2][4];
    {
        V8F z;
#pragma unroll
        for (int i = 0; i < 8; ++i) z.f[i] = 0.f;
#pragma unroll
        for (int mt = 0; mt < 2; ++mt)
#pragma unroll
            for (int nt = 0; nt < 4; ++nt) acc[mt][nt] = z.v;
    }

    // Per-thread chunk map: chunk c in [0,512): row r=c>>2, 16B segment seg=c&3.
    const int r0 = tid >> 2, seg0 = tid & 3;          // chunk tid
    const int r1 = r0 + 64,  seg1 = seg0;             // chunk tid+256

    // Issue one tile (A+B) into buffer `buf` for k-step kt: 4 async instrs/wave.
    auto issue_tile = [&](int kt, int buf) {
        const int kh = kt * 32;   // k offset in halves
        // A rows m0+r, BT rows n0+r
        async_copy_b128(smem_base + (unsigned)(((buf * 2 + 0) * 128 * LDA + r0 * LDA + seg0 * 8) * 2),
                        A + (size_t)(m0 + r0) * KDIM + kh + seg0 * 8);
        async_copy_b128(smem_base + (unsigned)(((buf * 2 + 0) * 128 * LDA + r1 * LDA + seg1 * 8) * 2),
                        A + (size_t)(m0 + r1) * KDIM + kh + seg1 * 8);
        async_copy_b128(smem_base + (unsigned)(((buf * 2 + 1) * 128 * LDA + r0 * LDA + seg0 * 8) * 2),
                        BT + (size_t)(n0 + r0) * KDIM + kh + seg0 * 8);
        async_copy_b128(smem_base + (unsigned)(((buf * 2 + 1) * 128 * LDA + r1 * LDA + seg1 * 8) * 2),
                        BT + (size_t)(n0 + r1) * KDIM + kh + seg1 * 8);
    };

    constexpr int KT = KDIM / 32;
    issue_tile(0, 0);

    for (int kt = 0; kt < KT; ++kt) {
        const int buf = kt & 1;
        if (kt + 1 < KT) {
            issue_tile(kt + 1, buf ^ 1);   // overlap next tile with this tile's math
            wait_async_le4();              // in-order: first 4 (tile kt) have landed
        } else {
            wait_async_0();
        }
        __syncthreads();                   // tile kt visible to all waves

        const _Float16* As = &smem[buf][0][0];
        const _Float16* Bs = &smem[buf][1][0];
        v16h af[2], bf[4];
#pragma unroll
        for (int mt = 0; mt < 2; ++mt)
            af[mt] = load_frag_rowk(As + (wm * 32 + mt * 16) * LDA, LDA, lane);
#pragma unroll
        for (int nt = 0; nt < 4; ++nt)
            bf[nt] = load_frag_rowk(Bs + (wn * 64 + nt * 16) * LDA, LDA, lane);
        __syncthreads();                   // reads done before buf is re-DMA'd

#pragma unroll
        for (int mt = 0; mt < 2; ++mt)
#pragma unroll
            for (int nt = 0; nt < 4; ++nt)
                acc[mt][nt] = wmma16(af[mt], bf[nt], acc[mt][nt]);
    }

    // Epilogue
    const int nloc = lane & 15;
    const int half = lane >> 4;
#pragma unroll
    for (int mt = 0; mt < 2; ++mt) {
#pragma unroll
        for (int nt = 0; nt < 4; ++nt) {
            V8F c; c.v = acc[mt][nt];
            const int n  = n0 + wn * 64 + nt * 16 + nloc;
            const float bv = bias[n];
#pragma unroll
            for (int rr = 0; rr < 8; ++rr) {
                const int m = m0 + wm * 32 + mt * 16 + rr + half * 8;
                float v = c.f[rr] + bv;
                if (EPI == 2) v = gelu_tanh(v);
                if (EPI == 1) {
                    v += resid[(size_t)m * N + n];
                    ((float*)outp)[(size_t)m * N + n] = v;
                } else {
                    ((_Float16*)outp)[(size_t)m * N + n] = (_Float16)v;
                }
            }
        }
    }
}

// ---------------------------------------------------------------------------
// Windowed attention: block per window (2048 blocks), wave per head (x2 heads).
// qkv: [ROWS][1536] f16 (Q|K|V), bias_full: [16][64][64] f32, o: [ROWS][512] f16
// ---------------------------------------------------------------------------
__global__ __launch_bounds__(256)
void attn_kernel(const _Float16* __restrict__ qkv, const float* __restrict__ bias_full,
                 _Float16* __restrict__ o) {
    __shared__ _Float16 Pbuf[8][64 * 64];   // 64 KB: per-wave softmax(P) staging
    const int wi   = blockIdx.x;
    const size_t row0 = (size_t)wi * 64;
    const int lane = threadIdx.x & 31;
    const int wave = threadIdx.x >> 5;
    const int nloc = lane & 15;
    const int half = lane >> 4;
    _Float16* P = Pbuf[wave];
    const float scale = 0.17677669529663687f;  // 1/sqrt(32)

    for (int hh = 0; hh < 2; ++hh) {
        const int head = wave + hh * 8;
        const _Float16* qbase = qkv + row0 * 1536 + head * HD;
        const _Float16* kbase = qbase + CC;
        const _Float16* vbase = qbase + 2 * CC;

        // V fragments: [key(kt*32..)][dim(nt*16..)] once per head
        v16h vf[2][2];
#pragma unroll
        for (int kt = 0; kt < 2; ++kt)
#pragma unroll
            for (int nt = 0; nt < 2; ++nt)
                vf[kt][nt] = load_frag_kn(vbase + (size_t)(kt * 32) * 1536 + nt * 16,
                                          1536, lane);

        // ---- scores + softmax, one 16-row strip at a time ----
        for (int mt = 0; mt < 4; ++mt) {
            v16h qa = load_frag_rowk(qbase + (size_t)(mt * 16) * 1536, 1536, lane);
            v8f sc[4];
            {
                V8F z;
#pragma unroll
                for (int i = 0; i < 8; ++i) z.f[i] = 0.f;
#pragma unroll
                for (int nt = 0; nt < 4; ++nt) {
                    v16h kb = load_frag_rowk(kbase + (size_t)(nt * 16) * 1536, 1536, lane);
                    sc[nt] = wmma16(qa, kb, z.v);
                }
            }
            float s[4][8];
#pragma unroll
            for (int nt = 0; nt < 4; ++nt) {
                V8F c; c.v = sc[nt];
#pragma unroll
                for (int rr = 0; rr < 8; ++rr) {
                    int m = mt * 16 + rr + half * 8;
                    s[nt][rr] = c.f[rr] * scale +
                                bias_full[head * 4096 + m * 64 + nt * 16 + nloc];
                }
            }
#pragma unroll
            for (int rr = 0; rr < 8; ++rr) {
                float mx = s[0][rr];
#pragma unroll
                for (int nt = 1; nt < 4; ++nt) mx = fmaxf(mx, s[nt][rr]);
#pragma unroll
                for (int msk = 1; msk < 16; msk <<= 1)
                    mx = fmaxf(mx, __shfl_xor(mx, msk, 32));
                float sum = 0.f;
#pragma unroll
                for (int nt = 0; nt < 4; ++nt) {
                    s[nt][rr] = __expf(s[nt][rr] - mx);
                    sum += s[nt][rr];
                }
#pragma unroll
                for (int msk = 1; msk < 16; msk <<= 1)
                    sum += __shfl_xor(sum, msk, 32);
                const float inv = 1.f / sum;
                const int m = mt * 16 + rr + half * 8;
#pragma unroll
                for (int nt = 0; nt < 4; ++nt)
                    P[m * 64 + nt * 16 + nloc] = (_Float16)(s[nt][rr] * inv);
            }
        }
        __syncthreads();   // P ready (uniform across waves)

        // ---- O = P @ V ----
        for (int mt = 0; mt < 4; ++mt) {
            v16h pa0 = load_frag_rowk(P + (mt * 16) * 64, 64, lane);
            v16h pa1 = load_frag_rowk(P + (mt * 16) * 64 + 32, 64, lane);
#pragma unroll
            for (int nt = 0; nt < 2; ++nt) {
                V8F z;
#pragma unroll
                for (int i = 0; i < 8; ++i) z.f[i] = 0.f;
                v8f oc = wmma16(pa0, vf[0][nt], z.v);
                oc = wmma16(pa1, vf[1][nt], oc);
                V8F c; c.v = oc;
#pragma unroll
                for (int rr = 0; rr < 8; ++rr) {
                    size_t m = row0 + mt * 16 + rr + half * 8;
                    o[m * CC + head * HD + nt * 16 + nloc] = (_Float16)c.f[rr];
                }
            }
        }
        __syncthreads();   // done with P before next head overwrites
    }
}

// ---------------------------------------------------------------------------
// Launcher
// ---------------------------------------------------------------------------
extern "C" void kernel_launch(void* const* d_in, const int* in_sizes, int n_in,
                              void* d_out, int out_size, void* d_ws, size_t ws_size,
                              hipStream_t stream) {
    (void)in_sizes; (void)n_in; (void)out_size; (void)ws_size;
    const float* x        = (const float*)d_in[0];
    const float* gamma1   = (const float*)d_in[1];
    const float* beta1    = (const float*)d_in[2];
    const float* gamma2   = (const float*)d_in[3];
    const float* beta2    = (const float*)d_in[4];
    const float* qkv_w    = (const float*)d_in[5];
    const float* qkv_b    = (const float*)d_in[6];
    const float* merge_w  = (const float*)d_in[7];
    const float* merge_b  = (const float*)d_in[8];
    const float* bias_tab = (const float*)d_in[9];
    const float* w1       = (const float*)d_in[10];
    const float* b1       = (const float*)d_in[11];
    const float* w2       = (const float*)d_in[12];
    const float* b2       = (const float*)d_in[13];
    const int*   rel_idx  = (const int*)d_in[14];

    char* ws = (char*)d_ws;
    float*    XW    = (float*)(ws);                          // 268435456 B
    _Float16* H16   = (_Float16*)(ws + 268435456ull);        // 134217728 B
    _Float16* QKV16 = (_Float16*)(ws + 402653184ull);        // 402653184 B
    _Float16* O16   = (_Float16*)(ws + 805306368ull);        // 134217728 B
    _Float16* MID16 = (_Float16*)(ws + 939524096ull);        // 536870912 B
    _Float16* QKVWT = (_Float16*)(ws + 1476395008ull);       //   1572864 B
    _Float16* MRGWT = (_Float16*)(ws + 1477967872ull);       //    524288 B
    _Float16* W1T   = (_Float16*)(ws + 1478492160ull);       //   2097152 B
    _Float16* W2T   = (_Float16*)(ws + 1480589312ull);       //   2097152 B
    float*    BIASF = (float*)(ws + 1482686464ull);          //    262144 B

    // Prep: weight f32->f16 transposes, dense bias table
    transpose_to_f16<<<(512 * 1536 + 255) / 256, 256, 0, stream>>>(qkv_w, QKVWT, 512, 1536);
    transpose_to_f16<<<(512 * 512 + 255) / 256, 256, 0, stream>>>(merge_w, MRGWT, 512, 512);
    transpose_to_f16<<<(512 * 2048 + 255) / 256, 256, 0, stream>>>(w1, W1T, 512, 2048);
    transpose_to_f16<<<(2048 * 512 + 255) / 256, 256, 0, stream>>>(w2, W2T, 2048, 512);
    build_bias<<<(NH * 64 * 64 + 255) / 256, 256, 0, stream>>>(bias_tab, rel_idx, BIASF);

    // Partition + LN1
    partition_kernel<<<(BB * CC * HH * WW) / 256, 256, 0, stream>>>(x, XW);
    ln_kernel<<<ROWS, 128, 0, stream>>>(XW, gamma1, beta1, H16);

    // QKV projection: [ROWS,512] x [512,1536] -> f16
    gemm_kernel<512, 0><<<dim3(1536 / 128, ROWS / 128), 256, 0, stream>>>(
        H16, QKVWT, qkv_b, nullptr, QKV16, 1536);

    // Windowed relative-position attention
    attn_kernel<<<BB * 64, 256, 0, stream>>>(QKV16, BIASF, O16);

    // Merge projection + residual (in-place into XW, f32)
    gemm_kernel<512, 1><<<dim3(512 / 128, ROWS / 128), 256, 0, stream>>>(
        O16, MRGWT, merge_b, XW, XW, 512);

    // LN2 + MLP
    ln_kernel<<<ROWS, 128, 0, stream>>>(XW, gamma2, beta2, H16);
    gemm_kernel<512, 2><<<dim3(2048 / 128, ROWS / 128), 256, 0, stream>>>(
        H16, W1T, b1, nullptr, MID16, 2048);
    gemm_kernel<2048, 1><<<dim3(512 / 128, ROWS / 128), 256, 0, stream>>>(
        MID16, W2T, b2, XW, XW, 512);

    // Departition to output
    departition_kernel<<<(BB * CC * HH * WW) / 256, 256, 0, stream>>>(XW, (float*)d_out);
}
